// ScatterHead_11965778886953
// MI455X (gfx1250) — compile-verified, matrix operand order checked by reading, not run on hardware
//
#include <hip/hip_runtime.h>
#include <hip/hip_bf16.h>

// ---------------------------------------------------------------------------
// CDNA5 (gfx1250) implementation: WMMA f16 GEMM core + persistent wave32
// LSTM recurrence + attention decoder. Weights converted once per launch to
// f16 in workspace; each f16 copy goes into one of 4 reused buffers (~5.4MB total).
// All GEMM M dims are multiples of 128 and K dims multiples of 32, so tile
// staging is fully vectorized and unguarded (only N is guarded, for N=96).
// ---------------------------------------------------------------------------

typedef __attribute__((ext_vector_type(16))) _Float16 v16h;
typedef __attribute__((ext_vector_type(8)))  _Float16 v8h;
typedef __attribute__((ext_vector_type(4)))  _Float16 v4h;
typedef __attribute__((ext_vector_type(8)))  float    v8f;
typedef __attribute__((ext_vector_type(4)))  float    v4f;

__device__ __forceinline__ v8f wmma32(v16h a, v16h b, v8f c) {
  // (neg_a, A, neg_b, B, c_mod, C, reuse_a, reuse_b)
  return __builtin_amdgcn_wmma_f32_16x16x32_f16(false, a, false, b,
                                                (short)0, c, false, false);
}

__device__ __forceinline__ float sigf(float x) {
  return 1.f / (1.f + __expf(-x));
}

// Load a 16x32 f16 fragment (A or B^T operand) for this lane.
// ISA 7.12.2: row = lane&15; lane-half selects K groups; VGPR v holds
// K = (v/4)*16 + (lane/16)*8 + (v%4)*2 , +1.  For fixed lane the 16 halves
// are two contiguous 16B runs -> compiler emits 2x b128 loads.
__device__ __forceinline__ v16h frag_ld16(const _Float16* __restrict__ p,
                                          int stride) {
  const int lane = threadIdx.x & 31;
  const int r = lane & 15;
  const int half = lane >> 4;
  v16h f;
#pragma unroll
  for (int v = 0; v < 8; ++v) {
    const int kb = ((v >> 2) << 4) + (half << 3) + ((v & 3) << 1);
    const _Float16* q = p + (size_t)r * stride + kb;
    f[2 * v]     = q[0];
    f[2 * v + 1] = q[1];
  }
  return f;
}

// ---------------------------------------------------------------------------
// f32 -> f16 weight conversion (vectorized; n is always a multiple of 4)
// ---------------------------------------------------------------------------
__global__ void cvt_f16_kernel(const float* __restrict__ s,
                               _Float16* __restrict__ d, int n4) {
  int i = blockIdx.x * blockDim.x + threadIdx.x;
  if (i >= n4) return;
  v4f a = *(const v4f*)&s[(size_t)i * 4];
  v4h h;
  h.x = (_Float16)a.x; h.y = (_Float16)a.y;
  h.z = (_Float16)a.z; h.w = (_Float16)a.w;
  *(v4h*)&d[(size_t)i * 4] = h;
}

// ---------------------------------------------------------------------------
// visual = mean over W of img[B,D,W,T] -> [B*T, D], LDS-tiled transpose so
// global reads are coalesced along T and writes coalesced along D.
// grid = (D/32, B), 256 threads.
// ---------------------------------------------------------------------------
__global__ __launch_bounds__(256) void mean_pool_kernel(
    const float* __restrict__ img, float* __restrict__ vis) {
  __shared__ float tile[32][65];
  const int b = blockIdx.y;
  const int d0 = blockIdx.x * 32;
  const int tid = threadIdx.x;
  const int t = tid & 63;
#pragma unroll
  for (int j = 0; j < 8; ++j) {
    const int dd = (tid >> 6) + j * 4;
    float s = 0.f;
#pragma unroll
    for (int w = 0; w < 8; ++w)
      s += img[(((size_t)b * 512 + d0 + dd) * 8 + w) * 64 + t];
    tile[dd][t] = s * 0.125f;
  }
  __syncthreads();
  const int dd = tid & 31;
#pragma unroll
  for (int j = 0; j < 8; ++j) {
    const int tt = (tid >> 5) + j * 8;
    vis[((size_t)b * 64 + tt) * 512 + d0 + dd] = tile[dd][tt];
  }
}

// fusion[:,0:512] = ctx ; fusion[:,512:1024] = visual  (float4 copies)
__global__ void concat_fusion_kernel(const float* __restrict__ ctx,
                                     const float* __restrict__ vis,
                                     float* __restrict__ fus) {
  int i = blockIdx.x * blockDim.x + threadIdx.x;   // over 8192*512/4
  if (i >= 8192 * 128) return;
  int m = i >> 7, q = (i & 127) * 4;
  v4f c = *(const v4f*)&ctx[(size_t)m * 512 + q];
  v4f v = *(const v4f*)&vis[(size_t)m * 512 + q];
  *(v4f*)&fus[(size_t)m * 1024 + q] = c;
  *(v4f*)&fus[(size_t)m * 1024 + 512 + q] = v;
}

// ---------------------------------------------------------------------------
// Generic WMMA GEMM:  C[m, coff+n] = (A[m,:K]f32 @ W[n,:K]f16^T) + bias[n]
//                     (optionally * mul[m*ldm+n])
// Requires M % 128 == 0 and K % 32 == 0 (true for every call here).
// Block tile 128(M) x 64(N), 8 waves in 4x2 layout, 32x32 per wave.
// ---------------------------------------------------------------------------
__global__ __launch_bounds__(256) void gemm_wmma_kernel(
    const float* __restrict__ A, int lda,
    const _Float16* __restrict__ W,
    const float* __restrict__ bias,
    const float* __restrict__ mul, int ldm,
    float* __restrict__ C, int ldc, int coff,
    int N, int K) {
  __shared__ _Float16 sA[128 * 32];
  __shared__ _Float16 sB[64 * 32];
  const int tid = threadIdx.x;
  const int lane = tid & 31, wid = tid >> 5;
  const int r0 = blockIdx.y * 128, n0 = blockIdx.x * 64;
  const int wm = (wid & 3) * 32, wn = (wid >> 2) * 32;
  // A staging: 4 float4 loads per thread (rows ar+32j, k quad ak)
  const int ar = tid >> 3;
  const int ak = (tid & 7) * 4;
  // W staging: one 16B (8-half) load per thread
  const int br = tid >> 2;
  const int bk = (tid & 3) * 8;
  const bool bok = (n0 + br) < N;
  const _Float16* wrow = &W[(size_t)(n0 + br) * K + bk];

  v8f z = {};
  v8f acc[2][2];
  acc[0][0] = z; acc[0][1] = z; acc[1][0] = z; acc[1][1] = z;

  for (int k0 = 0; k0 < K; k0 += 32) {
#pragma unroll
    for (int j = 0; j < 4; ++j) {
      const int r = ar + j * 32;
      v4f a4 = *(const v4f*)&A[(size_t)(r0 + r) * lda + k0 + ak];
      v4h h4;
      h4.x = (_Float16)a4.x; h4.y = (_Float16)a4.y;
      h4.z = (_Float16)a4.z; h4.w = (_Float16)a4.w;
      *(v4h*)&sA[r * 32 + ak] = h4;
    }
    v8h w8 = {};
    if (bok) w8 = *(const v8h*)&wrow[k0];
    *(v8h*)&sB[br * 32 + bk] = w8;
    if (k0 + 32 < K)  // gfx1250 global_prefetch_b8 on next A slab
      __builtin_prefetch(&A[(size_t)(r0 + ar) * lda + k0 + 32 + ak], 0, 1);
    __syncthreads();

    v16h a0 = frag_ld16(&sA[(wm + 0) * 32], 32);
    v16h a1 = frag_ld16(&sA[(wm + 16) * 32], 32);
    v16h b0 = frag_ld16(&sB[(wn + 0) * 32], 32);
    v16h b1 = frag_ld16(&sB[(wn + 16) * 32], 32);
    acc[0][0] = wmma32(a0, b0, acc[0][0]);
    acc[0][1] = wmma32(a0, b1, acc[0][1]);
    acc[1][0] = wmma32(a1, b0, acc[1][0]);
    acc[1][1] = wmma32(a1, b1, acc[1][1]);
    __syncthreads();
  }

  const int cn = lane & 15, chalf = lane >> 4;
#pragma unroll
  for (int ti = 0; ti < 2; ++ti)
#pragma unroll
    for (int tj = 0; tj < 2; ++tj)
#pragma unroll
      for (int v = 0; v < 8; ++v) {
        const int m = r0 + wm + ti * 16 + v + chalf * 8;
        const int n = n0 + wn + tj * 16 + cn;
        if (n < N) {
          float val = acc[ti][tj][v] + (bias ? bias[n] : 0.f);
          if (mul) val *= mul[(size_t)m * ldm + n];
          C[(size_t)m * ldc + coff + n] = val;
        }
      }
}

// ---------------------------------------------------------------------------
// Persistent LSTM recurrence. Grid = B/16 workgroups, 256 threads (8 waves).
// preG[b*T+t, 0:1024] already holds x@W_ih^T + b. Each step computes
// h@W_hh^T with WMMA (A from LDS f16 h, B streamed from L2-resident f16
// W_hh), applies the cell, writes h into hs[:, hoff:hoff+256] (stride ldh).
// Wave w owns hidden units [w*32, w*32+32) across all 4 gates.
// ---------------------------------------------------------------------------
__global__ __launch_bounds__(256) void lstm_rec_kernel(
    const float* __restrict__ preG,
    const _Float16* __restrict__ Whh,   // [1024, 256] f16
    float* __restrict__ hs, int ldh, int hoff,
    int reverse) {
  __shared__ _Float16 sH[16 * 256];
  const int tid = threadIdx.x, lane = tid & 31, wid = tid >> 5;
  const int b0 = blockIdx.x * 16;
  for (int i = tid; i < 16 * 256; i += 256) sH[i] = (_Float16)0.f;
  float c[16];
#pragma unroll
  for (int i = 0; i < 16; ++i) c[i] = 0.f;
  __syncthreads();

  const int cn = lane & 15, chalf = lane >> 4;
  v8f z = {};
  for (int tt = 0; tt < 64; ++tt) {
    const int t = reverse ? (63 - tt) : tt;
    v8f acc[4][2];
#pragma unroll
    for (int g = 0; g < 4; ++g) { acc[g][0] = z; acc[g][1] = z; }

    for (int kk = 0; kk < 8; ++kk) {
      const int k0 = kk * 32;
      v16h a = frag_ld16(&sH[k0], 256);
#pragma unroll
      for (int g = 0; g < 4; ++g)
#pragma unroll
        for (int h = 0; h < 2; ++h) {
          const int ncol = g * 256 + wid * 32 + h * 16;
          v16h b = frag_ld16(&Whh[(size_t)ncol * 256 + k0], 256);
          acc[g][h] = wmma32(a, b, acc[g][h]);
        }
    }
    __syncthreads();  // everyone done reading sH for this step

#pragma unroll
    for (int h = 0; h < 2; ++h) {
      const int j = wid * 32 + h * 16 + cn;
#pragma unroll
      for (int v = 0; v < 8; ++v) {
        const int m = v + chalf * 8;
        const size_t base = ((size_t)(b0 + m) * 64 + t) * 1024;
        float gi = sigf(preG[base + j]        + acc[0][h][v]);
        float gf = sigf(preG[base + 256 + j]  + acc[1][h][v]);
        float gg = tanhf(preG[base + 512 + j] + acc[2][h][v]);
        float go = sigf(preG[base + 768 + j]  + acc[3][h][v]);
        const int ci = h * 8 + v;
        c[ci] = gf * c[ci] + gi * gg;
        float hv = go * tanhf(c[ci]);
        hs[((size_t)(b0 + m) * 64 + t) * ldh + hoff + j] = hv;
        sH[m * 256 + j] = (_Float16)hv;
      }
    }
    __syncthreads();
  }
}

// ---------------------------------------------------------------------------
// Attention decoder. Grid = B/16 workgroups, 256 threads, loops S=51 steps.
// The one-hot decoder input contributes exactly one column of cell_wih
// (index F + text[b,s]) so no GEMM is needed for it.
// ---------------------------------------------------------------------------
__global__ __launch_bounds__(256) void attn_decoder_kernel(
    const float* __restrict__ proj,     // [B*T, 256]
    const float* __restrict__ Hf,       // [B*T, 1024] (batch_H)
    const _Float16* __restrict__ h2h,   // [256, 256]
    const float* __restrict__ h2h_b,    // [256]
    const float* __restrict__ score_w,  // [256]
    const _Float16* __restrict__ cwih,  // [1024, 1120]
    const _Float16* __restrict__ cwhh,  // [1024, 256]
    const float* __restrict__ cb,       // [1024]
    const int* __restrict__ text,       // [B, 51]
    float* __restrict__ hsdec)          // [B, 51, 256]
{
  __shared__ _Float16 sH[16 * 256];
  __shared__ _Float16 sCtx[16 * 1024];
  __shared__ float sHp[16 * 256];
  __shared__ float sE[16 * 64];
  __shared__ float sAl[16 * 64];
  __shared__ float sSv[256];
  const int tid = threadIdx.x, lane = tid & 31, wid = tid >> 5;
  const int b0 = blockIdx.x * 16;
  for (int i = tid; i < 16 * 256; i += 256) sH[i] = (_Float16)0.f;
  sSv[tid] = score_w[tid];
  float c[16];
#pragma unroll
  for (int i = 0; i < 16; ++i) c[i] = 0.f;
  __syncthreads();

  const int cn = lane & 15, chalf = lane >> 4;
  v8f z = {};
  for (int s = 0; s < 51; ++s) {
    // ---- 1. hp = h @ h2h^T + b  (wave w -> cols [w*32, w*32+32))
    v8f p0 = z, p1 = z;
    for (int kk = 0; kk < 8; ++kk) {
      const int k0 = kk * 32;
      v16h a = frag_ld16(&sH[k0], 256);
      v16h bA = frag_ld16(&h2h[(size_t)(wid * 32) * 256 + k0], 256);
      v16h bB = frag_ld16(&h2h[(size_t)(wid * 32 + 16) * 256 + k0], 256);
      p0 = wmma32(a, bA, p0);
      p1 = wmma32(a, bB, p1);
    }
#pragma unroll
    for (int v = 0; v < 8; ++v) {
      const int m = v + chalf * 8;
      const int n = wid * 32 + cn;
      sHp[m * 256 + n]      = p0[v] + h2h_b[n];
      sHp[m * 256 + n + 16] = p1[v] + h2h_b[n + 16];
    }
    __syncthreads();

    // ---- 2. energies e[m,t] = tanh(proj + hp) . sv
#pragma unroll 1
    for (int r = 0; r < 4; ++r) {
      const int p = tid + 256 * r;
      const int m = p >> 6, t = p & 63;
      const float* pr = &proj[((size_t)(b0 + m) * 64 + t) * 256];
      const float* hp = &sHp[m * 256];
      float av = 0.f;
      for (int kh = 0; kh < 256; kh += 4) {
        v4f p4 = *(const v4f*)&pr[kh];
        av += tanhf(p4.x + hp[kh])     * sSv[kh];
        av += tanhf(p4.y + hp[kh + 1]) * sSv[kh + 1];
        av += tanhf(p4.z + hp[kh + 2]) * sSv[kh + 2];
        av += tanhf(p4.w + hp[kh + 3]) * sSv[kh + 3];
      }
      sE[m * 64 + t] = av;
    }
    __syncthreads();

    // ---- 3. softmax over T per row
    if (tid < 16) {
      float mx = -1e30f;
      for (int t = 0; t < 64; ++t) mx = fmaxf(mx, sE[tid * 64 + t]);
      float sum = 0.f;
      for (int t = 0; t < 64; ++t) {
        float e = __expf(sE[tid * 64 + t] - mx);
        sAl[tid * 64 + t] = e;
        sum += e;
      }
      float inv = 1.f / sum;
      for (int t = 0; t < 64; ++t) sAl[tid * 64 + t] *= inv;
    }
    __syncthreads();

    // ---- 4. ctx[m,f] = sum_t alpha[m,t] * Hf[b,t,f] (thread owns col tid+fo*256)
#pragma unroll 1
    for (int m = 0; m < 16; ++m) {
      const float* hr = &Hf[((size_t)(b0 + m) * 64) * 1024 + tid];
#pragma unroll 1
      for (int fo = 0; fo < 4; ++fo) {
        float av = 0.f;
        for (int t = 0; t < 64; ++t)
          av += sAl[m * 64 + t] * hr[(size_t)t * 1024 + fo * 256];
        sCtx[m * 1024 + fo * 256 + tid] = (_Float16)av;
      }
    }
    __syncthreads();

    // ---- 5. gates = ctx@cwih[:, :1024]^T + h@cwhh^T (+ one-hot column + b)
    v8f acc[4][2];
#pragma unroll
    for (int g = 0; g < 4; ++g) { acc[g][0] = z; acc[g][1] = z; }
    for (int kk = 0; kk < 32; ++kk) {
      const int k0 = kk * 32;
      v16h a = frag_ld16(&sCtx[k0], 1024);
#pragma unroll
      for (int g = 0; g < 4; ++g)
#pragma unroll
        for (int h = 0; h < 2; ++h) {
          const int ncol = g * 256 + wid * 32 + h * 16;
          v16h b = frag_ld16(&cwih[(size_t)ncol * 1120 + k0], 1120);
          acc[g][h] = wmma32(a, b, acc[g][h]);
        }
    }
    for (int kk = 0; kk < 8; ++kk) {
      const int k0 = kk * 32;
      v16h a = frag_ld16(&sH[k0], 256);
#pragma unroll
      for (int g = 0; g < 4; ++g)
#pragma unroll
        for (int h = 0; h < 2; ++h) {
          const int ncol = g * 256 + wid * 32 + h * 16;
          v16h b = frag_ld16(&cwhh[(size_t)ncol * 256 + k0], 256);
          acc[g][h] = wmma32(a, b, acc[g][h]);
        }
    }
    __syncthreads();  // done reading sH / sCtx

#pragma unroll
    for (int h = 0; h < 2; ++h) {
      const int j = wid * 32 + h * 16 + cn;
#pragma unroll
      for (int v = 0; v < 8; ++v) {
        const int m = v + chalf * 8;
        const int idx = text[(b0 + m) * 51 + s];
        float oh0 = (float)cwih[(size_t)(0 * 256 + j) * 1120 + 1024 + idx];
        float oh1 = (float)cwih[(size_t)(1 * 256 + j) * 1120 + 1024 + idx];
        float oh2 = (float)cwih[(size_t)(2 * 256 + j) * 1120 + 1024 + idx];
        float oh3 = (float)cwih[(size_t)(3 * 256 + j) * 1120 + 1024 + idx];
        float gi = sigf(cb[j] + oh0 + acc[0][h][v]);
        float gf = sigf(cb[256 + j] + oh1 + acc[1][h][v]);
        float gg = tanhf(cb[512 + j] + oh2 + acc[2][h][v]);
        float go = sigf(cb[768 + j] + oh3 + acc[3][h][v]);
        const int ci = h * 8 + v;
        c[ci] = gf * c[ci] + gi * gg;
        float hv = go * tanhf(c[ci]);
        hsdec[((size_t)(b0 + m) * 51 + s) * 256 + j] = hv;
        sH[m * 256 + j] = (_Float16)hv;
      }
    }
    __syncthreads();
  }
}

// ---------------------------------------------------------------------------
// Host orchestration
// ---------------------------------------------------------------------------
extern "C" void kernel_launch(void* const* d_in, const int* in_sizes, int n_in,
                              void* d_out, int out_size, void* d_ws,
                              size_t ws_size, hipStream_t stream) {
  (void)in_sizes; (void)n_in; (void)out_size; (void)ws_size;
  const float* l1_wih_f = (const float*)d_in[0];
  const float* l1_whh_f = (const float*)d_in[1];
  const float* l1_b_f   = (const float*)d_in[2];
  const float* l1_wih_b = (const float*)d_in[3];
  const float* l1_whh_b = (const float*)d_in[4];
  const float* l1_b_b   = (const float*)d_in[5];
  const float* l1_ow    = (const float*)d_in[6];
  const float* l1_ob    = (const float*)d_in[7];
  const float* l2_wih_f = (const float*)d_in[8];
  const float* l2_whh_f = (const float*)d_in[9];
  const float* l2_b_f   = (const float*)d_in[10];
  const float* l2_wih_b = (const float*)d_in[11];
  const float* l2_whh_b = (const float*)d_in[12];
  const float* l2_b_b   = (const float*)d_in[13];
  const float* l2_ow    = (const float*)d_in[14];
  const float* l2_ob    = (const float*)d_in[15];
  const float* sam_w    = (const float*)d_in[16];
  const float* sam_b    = (const float*)d_in[17];
  const float* i2h_w    = (const float*)d_in[18];
  const float* i2h_b    = (const float*)d_in[19];
  const float* h2h_w    = (const float*)d_in[20];
  const float* h2h_b    = (const float*)d_in[21];
  const float* score_w  = (const float*)d_in[22];
  const float* cell_wih = (const float*)d_in[23];
  const float* cell_whh = (const float*)d_in[24];
  const float* cell_b   = (const float*)d_in[25];
  const float* gen_w    = (const float*)d_in[26];
  const float* gen_b    = (const float*)d_in[27];
  const float* img      = (const float*)d_in[28];
  const int*   text     = (const int*)d_in[29];
  float* out = (float*)d_out;

  char* wsb = (char*)d_ws;
  size_t off = 0;
  auto alloc = [&](size_t bytes) -> void* {
    void* p = wsb + off;
    off += (bytes + 255) & ~(size_t)255;
    return p;
  };
  float* visual = (float*)alloc(8192ull * 512 * 4);
  float* fusion = (float*)alloc(8192ull * 1024 * 4);
  float* hcat   = (float*)alloc(8192ull * 512 * 4);
  float* preX   = (float*)alloc(8192ull * 1024 * 4);   // also SAM output
  float* seq    = (float*)alloc(8192ull * 256 * 4);
  float* ctxb   = (float*)alloc(8192ull * 512 * 4);
  float* proj   = (float*)alloc(8192ull * 256 * 4);
  float* hsdec  = (float*)alloc(6528ull * 256 * 4);
  _Float16* wA = (_Float16*)alloc(1146880ull * 2);     // wih / cell_wih / i2h
  _Float16* wC = (_Float16*)alloc(262144ull * 2);      // whh_f / h2h
  _Float16* wD = (_Float16*)alloc(262144ull * 2);      // whh_b / cell_whh
  _Float16* wE = (_Float16*)alloc(1048576ull * 2);     // ow / sam / gen

  auto cvt = [&](const float* s, _Float16* d, int n) {
    int n4 = n / 4;
    cvt_f16_kernel<<<(n4 + 255) / 256, 256, 0, stream>>>(s, d, n4);
  };
  auto gemm = [&](const float* A, int lda, const _Float16* W,
                  const float* bias, const float* mul, int ldm, float* C,
                  int ldc, int coff, int M, int N, int K) {
    dim3 g((N + 63) / 64, M / 128);
    gemm_wmma_kernel<<<g, 256, 0, stream>>>(A, lda, W, bias, mul, ldm, C,
                                            ldc, coff, N, K);
  };

  mean_pool_kernel<<<dim3(16, 128), 256, 0, stream>>>(img, visual);

  for (int i = 0; i < 5; ++i) {
    const float* xin = (i == 0) ? visual : ctxb;
    // ---------------- L1 BiLSTM (input width 512) ----------------
    cvt(l1_wih_f + (size_t)i * 1024 * 512, wA, 1024 * 512);
    gemm(xin, 512, wA, l1_b_f + (size_t)i * 1024, nullptr, 0, preX, 1024, 0,
         8192, 1024, 512);
    cvt(l1_whh_f + (size_t)i * 1024 * 256, wC, 1024 * 256);
    lstm_rec_kernel<<<8, 256, 0, stream>>>(preX, wC, hcat, 512, 0, 0);
    cvt(l1_wih_b + (size_t)i * 1024 * 512, wA, 1024 * 512);
    gemm(xin, 512, wA, l1_b_b + (size_t)i * 1024, nullptr, 0, preX, 1024, 0,
         8192, 1024, 512);
    cvt(l1_whh_b + (size_t)i * 1024 * 256, wD, 1024 * 256);
    lstm_rec_kernel<<<8, 256, 0, stream>>>(preX, wD, hcat, 512, 256, 1);
    cvt(l1_ow + (size_t)i * 256 * 512, wE, 256 * 512);
    gemm(hcat, 512, wE, l1_ob + (size_t)i * 256, nullptr, 0, seq, 256, 0,
         8192, 256, 512);
    // ---------------- L2 BiLSTM (input width 256) ----------------
    cvt(l2_wih_f + (size_t)i * 1024 * 256, wA, 1024 * 256);
    gemm(seq, 256, wA, l2_b_f + (size_t)i * 1024, nullptr, 0, preX, 1024, 0,
         8192, 1024, 256);
    cvt(l2_whh_f + (size_t)i * 1024 * 256, wC, 1024 * 256);
    lstm_rec_kernel<<<8, 256, 0, stream>>>(preX, wC, hcat, 512, 0, 0);
    cvt(l2_wih_b + (size_t)i * 1024 * 256, wA, 1024 * 256);
    gemm(seq, 256, wA, l2_b_b + (size_t)i * 1024, nullptr, 0, preX, 1024, 0,
         8192, 1024, 256);
    cvt(l2_whh_b + (size_t)i * 1024 * 256, wD, 1024 * 256);
    lstm_rec_kernel<<<8, 256, 0, stream>>>(preX, wD, hcat, 512, 256, 1);
    cvt(l2_ow + (size_t)i * 512 * 512, wE, 512 * 512);
    gemm(hcat, 512, wE, l2_ob + (size_t)i * 512, nullptr, 0, ctxb, 512, 0,
         8192, 512, 512);
    // ---------------- fusion + SAM gating ----------------
    concat_fusion_kernel<<<(8192 * 128 + 255) / 256, 256, 0, stream>>>(
        ctxb, visual, fusion);
    cvt(sam_w + (size_t)i * 1024 * 1024, wE, 1024 * 1024);
    gemm(fusion, 1024, wE, sam_b + (size_t)i * 1024, fusion, 1024, preX,
         1024, 0, 8192, 1024, 1024);        // preX <- batch_H (post-SAM)
    // ---------------- attention decode ----------------
    cvt(i2h_w + (size_t)i * 256 * 1024, wA, 256 * 1024);
    gemm(preX, 1024, wA, i2h_b + (size_t)i * 256, nullptr, 0, proj, 256, 0,
         8192, 256, 1024);
    cvt(h2h_w + (size_t)i * 256 * 256, wC, 256 * 256);
    cvt(cell_wih + (size_t)i * 1024 * 1120, wA, 1024 * 1120);
    cvt(cell_whh + (size_t)i * 1024 * 256, wD, 1024 * 256);
    attn_decoder_kernel<<<8, 256, 0, stream>>>(
        proj, preX, wC, h2h_b + (size_t)i * 256, score_w + (size_t)i * 256,
        wA, wD, cell_b + (size_t)i * 1024, text, hsdec);
    cvt(gen_w + (size_t)i * 96 * 256, wE, 96 * 256);
    gemm(hsdec, 256, wE, gen_b + (size_t)i * 96, nullptr, 0,
         out + (size_t)i * 128 * 51 * 96, 96, 0, 6528, 96, 256);
  }
}
